// MultiHeadAttention_42288247997025
// MI455X (gfx1250) — compile-verified, matrix operand order checked by reading, not run on hardware
//
#include <hip/hip_runtime.h>

// ---------------------------------------------------------------------------
// MHA forward for MI455X (gfx1250, wave32, WMMA + async-to-LDS staging).
// All GEMMs run on v_wmma_f32_16x16x32_bf16 (f32 accumulate).
// d_out = [ out (2,2048,1024) f32 | present_k (2,16,2048,64) f32 | present_v ]
// d_ws  = [ Wq|Wk|Wv|Wo bf16 (8MB) | q_bf 8MB | k_bf 8MB | vT_bf 8MB | attn_bf 8MB ]
// ---------------------------------------------------------------------------

typedef __attribute__((ext_vector_type(16))) __bf16 v16bf;
typedef __attribute__((ext_vector_type(8)))  __bf16 v8bf;
typedef __attribute__((ext_vector_type(4)))  __bf16 v4bf;
typedef __attribute__((ext_vector_type(8)))  float  v8f;
typedef __attribute__((ext_vector_type(4)))  float  v4f;

#define D_MODEL 1024
#define NSEQ    2048
#define BATCH   2
#define NHEAD   16
#define DHEAD   64

// LDS layout for k_attn (bytes), 64-bank-conflict-free padded strides:
//   K chunk  : 32 rows x 72 bf16  (144B stride)  = 4608B
//   V chunk  : 64 rows x 40 bf16  ( 80B stride)  = 5120B
//   kv buffer = 9728B, double buffered at 0 / 9728
//   P slices : 8 waves x (16 x 40 bf16 = 1280B)  at 19456
//   out stage: 8 waves x (16 x 68 f32 = 4352B, stride 4480) overlaid at 0
#define KV_BUF   9728
#define K_STRIDE 72
#define V_OFF    4608
#define V_STRIDE 40
#define P_OFF    19456
#define P_SLICE  1280
#define P_STRIDE 40
#define O_SLICE  4480
#define O_STRIDE 68
#define SMEM_BYTES 35840

// ---- fragment helpers ------------------------------------------------------
// A-fragment (16x32 bf16): lane = (hv,r); row M = r; element i holds
// K = (i/8)*16 + hv*8 + (i%8)  -> two contiguous 8-element (16B) runs.
// B-fragment (32x16 bf16): same packing, lane r = column N (K-major source).

__device__ __forceinline__ v16bf join16(v8bf lo, v8bf hi) {
  v16bf r;
#pragma unroll
  for (int i = 0; i < 8; ++i) { r[i] = lo[i]; r[i + 8] = hi[i]; }
  return r;
}

__device__ __forceinline__ v16bf load_frag_bf(const __bf16* row, int k0, int hv) {
  v8bf lo = *(const v8bf*)(row + k0 + hv * 8);
  v8bf hi = *(const v8bf*)(row + k0 + 16 + hv * 8);
  return join16(lo, hi);
}

__device__ __forceinline__ v16bf load_frag_f32(const float* row, int k0, int hv) {
  v8f lo = *(const v8f*)(row + k0 + hv * 8);
  v8f hi = *(const v8f*)(row + k0 + 16 + hv * 8);
  v16bf r;
#pragma unroll
  for (int i = 0; i < 8; ++i) { r[i] = (__bf16)lo[i]; r[i + 8] = (__bf16)hi[i]; }
  return r;
}

__device__ __forceinline__ v8f wmma_bf16(v16bf a, v16bf b, v8f c) {
  return __builtin_amdgcn_wmma_f32_16x16x32_bf16(false, a, false, b, (short)0, c,
                                                 false, false);
}

// async 16B global->LDS copy (ASYNCcnt-tracked DMA, gfx1250)
__device__ __forceinline__ void async_b128(unsigned lds_byte_off,
                                           const void* gaddr) {
  unsigned long long g = (unsigned long long)(size_t)gaddr;
  asm volatile("global_load_async_to_lds_b128 %0, %1, off" ::"v"(lds_byte_off),
               "v"(g)
               : "memory");
}

// ---- kernel 1: f32 -> bf16 conversion --------------------------------------
__global__ void k_cvt(const float* __restrict__ s, __bf16* __restrict__ d, int n4) {
  int i = blockIdx.x * blockDim.x + threadIdx.x;
  if (i >= n4) return;
  v4f x = ((const v4f*)s)[i];
  v4bf y;
#pragma unroll
  for (int j = 0; j < 4; ++j) y[j] = (__bf16)x[j];
  ((v4bf*)d)[i] = y;
}

// ---- kernel 2: Q/K/V projections (z selects which) -------------------------
__global__ __launch_bounds__(256) void k_proj(
    const float* __restrict__ Xq, const float* __restrict__ Xk,
    const float* __restrict__ Xv, const __bf16* __restrict__ wWq,
    const __bf16* __restrict__ wWk, const __bf16* __restrict__ wWv,
    const float* __restrict__ bq, const float* __restrict__ bk,
    const float* __restrict__ bv, __bf16* __restrict__ q_bf,
    float* __restrict__ k_f32, __bf16* __restrict__ k_bf,
    float* __restrict__ v_f32, __bf16* __restrict__ vT_bf) {
  const int z = blockIdx.z;
  const float*  X    = (z == 0) ? Xq : (z == 1) ? Xk : Xv;
  const __bf16* W    = (z == 0) ? wWq : (z == 1) ? wWk : wWv;
  const float*  bias = (z == 0) ? bq : (z == 1) ? bk : bv;

  const int tid = threadIdx.x;
  const int lane = tid & 31, wave = tid >> 5;
  const int r = lane & 15, hv = lane >> 4;
  const int M0 = blockIdx.y * 128 + (wave & 3) * 32;
  const int N0 = blockIdx.x * 128 + (wave >> 2) * 64;

  v8f acc[2][4];
#pragma unroll
  for (int mi = 0; mi < 2; ++mi)
#pragma unroll
    for (int nf = 0; nf < 4; ++nf)
#pragma unroll
      for (int j = 0; j < 8; ++j) acc[mi][nf][j] = 0.0f;

  const float* arow0 = X + (size_t)(M0 + r) * D_MODEL;
  const float* arow1 = X + (size_t)(M0 + 16 + r) * D_MODEL;
  const __bf16* brow[4];
#pragma unroll
  for (int nf = 0; nf < 4; ++nf) brow[nf] = W + (size_t)(N0 + 16 * nf + r) * D_MODEL;

  for (int k0 = 0; k0 < D_MODEL; k0 += 32) {
    v16bf a0 = load_frag_f32(arow0, k0, hv);
    v16bf a1 = load_frag_f32(arow1, k0, hv);
    v16bf bf[4];
#pragma unroll
    for (int nf = 0; nf < 4; ++nf) bf[nf] = load_frag_bf(brow[nf], k0, hv);
#pragma unroll
    for (int nf = 0; nf < 4; ++nf) {
      acc[0][nf] = wmma_bf16(a0, bf[nf], acc[0][nf]);
      acc[1][nf] = wmma_bf16(a1, bf[nf], acc[1][nf]);
    }
  }

#pragma unroll
  for (int mi = 0; mi < 2; ++mi) {
    const int tb = M0 + 16 * mi + 8 * hv;  // 8 consecutive tokens tb..tb+7
    const int bI = tb >> 11;               // batch
    const int nI = tb & 2047;              // token within batch
#pragma unroll
    for (int nf = 0; nf < 4; ++nf) {
      const int c = N0 + 16 * nf + r;
      const int h = c >> 6, dh = c & 63;
      const float bia = bias[c];
      const size_t base = (((size_t)bI * NHEAD + h) * NSEQ + nI) * DHEAD + dh;
      if (z == 0) {
#pragma unroll
        for (int j = 0; j < 8; ++j)
          q_bf[base + (size_t)j * DHEAD] = (__bf16)(acc[mi][nf][j] + bia);
      } else if (z == 1) {
#pragma unroll
        for (int j = 0; j < 8; ++j) {
          float val = acc[mi][nf][j] + bia;
          k_f32[base + (size_t)j * DHEAD] = val;
          k_bf[base + (size_t)j * DHEAD]  = (__bf16)val;
        }
      } else {
        v8bf pack;
#pragma unroll
        for (int j = 0; j < 8; ++j) {
          float val = acc[mi][nf][j] + bia;
          v_f32[base + (size_t)j * DHEAD] = val;
          pack[j] = (__bf16)val;
        }
        *(v8bf*)(vT_bf + (((size_t)bI * NHEAD + h) * DHEAD + dh) * NSEQ + nI) = pack;
      }
    }
  }
}

// ---- kernel 3: streaming attention with softmax-one ------------------------
// 8 waves x 16-query tiles; shared 32-key K/V chunks staged by async DMA,
// double buffered; out = acc / (1 + sum).
__global__ __launch_bounds__(256) void k_attn(
    const __bf16* __restrict__ q_bf, const __bf16* __restrict__ k_bf,
    const __bf16* __restrict__ vT_bf, __bf16* __restrict__ attn_bf) {
  __shared__ __align__(16) unsigned char smem[SMEM_BYTES];
  const int tid = threadIdx.x;
  const int lane = tid & 31, wave = tid >> 5;
  const int r = lane & 15, hv = lane >> 4;
  const int bh = blockIdx.y;
  const int b = bh >> 4, h = bh & 15;
  const int qt0 = blockIdx.x * 128 + wave * 16;

  const __bf16* qbase = q_bf + (size_t)bh * NSEQ * DHEAD;
  const __bf16* kbase = k_bf + (size_t)bh * NSEQ * DHEAD;
  const __bf16* vbase = vT_bf + (size_t)bh * DHEAD * NSEQ;

  const __bf16* qrow = qbase + (size_t)(qt0 + r) * DHEAD;
  const v16bf qa0 = load_frag_bf(qrow, 0, hv);
  const v16bf qa1 = load_frag_bf(qrow, 32, hv);

  // per-thread static pieces of the cooperative async copy
  const unsigned smem0 = (unsigned)(size_t)(void*)smem;
  const unsigned kRow = tid >> 3, kPiece = tid & 7;  // K: 32 rows x 8x16B
  const unsigned vRow = tid >> 2, vPiece = tid & 3;  // V: 64 rows x 4x16B
  const unsigned kLds = kRow * (K_STRIDE * 2) + kPiece * 16;
  const unsigned vLds = V_OFF + vRow * (V_STRIDE * 2) + vPiece * 16;
  const char* kG = (const char*)kbase + tid * 16;                  // +kb*128
  const char* vG = (const char*)(vbase + (size_t)vRow * NSEQ) + vPiece * 16;  // +kb*2

  v8f acc[4];
#pragma unroll
  for (int f = 0; f < 4; ++f)
#pragma unroll
    for (int j = 0; j < 8; ++j) acc[f][j] = 0.0f;
  float mrun[8], srun[8];
#pragma unroll
  for (int j = 0; j < 8; ++j) { mrun[j] = -1e30f; srun[j] = 0.0f; }

  __bf16* pP = (__bf16*)(smem + P_OFF + wave * P_SLICE);  // padded 16x32 P tile

  // prime buffer 0
  async_b128(smem0 + kLds, kG);
  async_b128(smem0 + vLds, vG);
  asm volatile("s_wait_asynccnt 0x0" ::: "memory");
  __syncthreads();

  for (int kc = 0; kc < NSEQ / 32; ++kc) {
    const int cur = kc & 1;
    if (kc + 1 < NSEQ / 32) {  // prefetch next chunk into the other buffer
      const int kb = (kc + 1) * 32;
      async_b128(smem0 + (cur ^ 1) * KV_BUF + kLds, kG + (size_t)kb * 128);
      async_b128(smem0 + (cur ^ 1) * KV_BUF + vLds, vG + (size_t)kb * 2);
    }

    const __bf16* kch = (const __bf16*)(smem + cur * KV_BUF);
    const __bf16* vch = (const __bf16*)(smem + cur * KV_BUF + V_OFF);

    // S(16x32) = q(16x64) * K^T : 4 WMMAs fed from LDS
    v8f s0, s1;
#pragma unroll
    for (int j = 0; j < 8; ++j) { s0[j] = 0.0f; s1[j] = 0.0f; }
    {
      const __bf16* kr0 = kch + (size_t)r * K_STRIDE;
      const __bf16* kr1 = kch + (size_t)(16 + r) * K_STRIDE;
      s0 = wmma_bf16(qa0, load_frag_bf(kr0, 0, hv), s0);
      s0 = wmma_bf16(qa1, load_frag_bf(kr0, 32, hv), s0);
      s1 = wmma_bf16(qa0, load_frag_bf(kr1, 0, hv), s1);
      s1 = wmma_bf16(qa1, load_frag_bf(kr1, 32, hv), s1);
    }
    // online softmax (row stats across one 16-lane half-wave)
    float rmax[8];
#pragma unroll
    for (int j = 0; j < 8; ++j) {
      s0[j] *= 0.125f;  // 1/sqrt(64)
      s1[j] *= 0.125f;
      rmax[j] = fmaxf(s0[j], s1[j]);
    }
#pragma unroll
    for (int off = 1; off < 16; off <<= 1)
#pragma unroll
      for (int j = 0; j < 8; ++j)
        rmax[j] = fmaxf(rmax[j], __shfl_xor(rmax[j], off, 32));

    float e0[8], e1[8], rsum[8], resc[8];
#pragma unroll
    for (int j = 0; j < 8; ++j) {
      float mnew = fmaxf(mrun[j], rmax[j]);
      resc[j] = __expf(mrun[j] - mnew);
      mrun[j] = mnew;
      e0[j] = __expf(s0[j] - mnew);
      e1[j] = __expf(s1[j] - mnew);
      rsum[j] = e0[j] + e1[j];
    }
#pragma unroll
    for (int off = 1; off < 16; off <<= 1)
#pragma unroll
      for (int j = 0; j < 8; ++j) rsum[j] += __shfl_xor(rsum[j], off, 32);
#pragma unroll
    for (int j = 0; j < 8; ++j) srun[j] = srun[j] * resc[j] + rsum[j];
#pragma unroll
    for (int f = 0; f < 4; ++f)
#pragma unroll
      for (int j = 0; j < 8; ++j) acc[f][j] *= resc[j];

    // P: C-layout -> A-layout via private padded LDS slice (wave-synchronous)
#pragma unroll
    for (int j = 0; j < 8; ++j) {
      pP[(j + 8 * hv) * P_STRIDE + r]      = (__bf16)e0[j];
      pP[(j + 8 * hv) * P_STRIDE + 16 + r] = (__bf16)e1[j];
    }
    asm volatile("s_wait_dscnt 0x0" ::: "memory");
    v16bf pa = load_frag_bf(pP + r * P_STRIDE, 0, hv);

    // acc(16x64) += P(16x32) * V(32x64): 4 WMMAs from transposed V in LDS
#pragma unroll
    for (int f = 0; f < 4; ++f)
      acc[f] = wmma_bf16(pa, load_frag_bf(vch + (size_t)(16 * f + r) * V_STRIDE, 0, hv),
                         acc[f]);

    // my async DMAs done + everyone finished reading this buffer
    asm volatile("s_wait_asynccnt 0x0" ::: "memory");
    __syncthreads();
  }

  // softmax-one: denominator = 1 + sum(exp(x - m))
#pragma unroll
  for (int j = 0; j < 8; ++j) {
    float inv = 1.0f / (1.0f + srun[j]);
#pragma unroll
    for (int f = 0; f < 4; ++f) acc[f][j] *= inv;
  }

  // stage padded 16x64 f32 tile in LDS (conflict-free), emit bf16 rows
  float* pO = (float*)(smem + wave * O_SLICE);
#pragma unroll
  for (int f = 0; f < 4; ++f)
#pragma unroll
    for (int j = 0; j < 8; ++j) pO[(j + 8 * hv) * O_STRIDE + 16 * f + r] = acc[f][j];
  asm volatile("s_wait_dscnt 0x0" ::: "memory");

  const int row = lane & 15, cb = (lane >> 4) * 32;
  __bf16* dst = attn_bf + ((size_t)b * NSEQ + qt0 + row) * D_MODEL + h * DHEAD + cb;
#pragma unroll
  for (int q = 0; q < 4; ++q) {
    v8bf o;
#pragma unroll
    for (int j = 0; j < 8; ++j) o[j] = (__bf16)pO[row * O_STRIDE + cb + q * 8 + j];
    *(v8bf*)(dst + q * 8) = o;
  }
}

// ---- kernel 4: output projection -------------------------------------------
__global__ __launch_bounds__(256) void k_out(const __bf16* __restrict__ attn_bf,
                                             const __bf16* __restrict__ wWo,
                                             const float* __restrict__ bo,
                                             float* __restrict__ out) {
  const int tid = threadIdx.x;
  const int lane = tid & 31, wave = tid >> 5;
  const int r = lane & 15, hv = lane >> 4;
  const int M0 = blockIdx.y * 128 + (wave & 3) * 32;
  const int N0 = blockIdx.x * 128 + (wave >> 2) * 64;

  v8f acc[2][4];
#pragma unroll
  for (int mi = 0; mi < 2; ++mi)
#pragma unroll
    for (int nf = 0; nf < 4; ++nf)
#pragma unroll
      for (int j = 0; j < 8; ++j) acc[mi][nf][j] = 0.0f;

  const __bf16* arow0 = attn_bf + (size_t)(M0 + r) * D_MODEL;
  const __bf16* arow1 = attn_bf + (size_t)(M0 + 16 + r) * D_MODEL;
  const __bf16* brow[4];
#pragma unroll
  for (int nf = 0; nf < 4; ++nf) brow[nf] = wWo + (size_t)(N0 + 16 * nf + r) * D_MODEL;

  for (int k0 = 0; k0 < D_MODEL; k0 += 32) {
    v16bf a0 = load_frag_bf(arow0, k0, hv);
    v16bf a1 = load_frag_bf(arow1, k0, hv);
    v16bf bf[4];
#pragma unroll
    for (int nf = 0; nf < 4; ++nf) bf[nf] = load_frag_bf(brow[nf], k0, hv);
#pragma unroll
    for (int nf = 0; nf < 4; ++nf) {
      acc[0][nf] = wmma_bf16(a0, bf[nf], acc[0][nf]);
      acc[1][nf] = wmma_bf16(a1, bf[nf], acc[1][nf]);
    }
  }

#pragma unroll
  for (int mi = 0; mi < 2; ++mi) {
    const int rb = M0 + 16 * mi + 8 * hv;
#pragma unroll
    for (int nf = 0; nf < 4; ++nf) {
      const int c = N0 + 16 * nf + r;
      const float bia = bo[c];
#pragma unroll
      for (int j = 0; j < 8; ++j)
        out[(size_t)(rb + j) * D_MODEL + c] = acc[mi][nf][j] + bia;
    }
  }
}

// ---- host launch -----------------------------------------------------------
extern "C" void kernel_launch(void* const* d_in, const int* in_sizes, int n_in,
                              void* d_out, int out_size, void* d_ws, size_t ws_size,
                              hipStream_t stream) {
  (void)in_sizes; (void)n_in; (void)out_size; (void)ws_size;
  const float* Q  = (const float*)d_in[0];
  const float* K  = (const float*)d_in[1];
  const float* V  = (const float*)d_in[2];
  const float* Wq = (const float*)d_in[3];
  const float* bq = (const float*)d_in[4];
  const float* Wk = (const float*)d_in[5];
  const float* bk = (const float*)d_in[6];
  const float* Wv = (const float*)d_in[7];
  const float* bv = (const float*)d_in[8];
  const float* Wo = (const float*)d_in[9];
  const float* bo = (const float*)d_in[10];

  float* out    = (float*)d_out;
  float* pres_k = out + (size_t)BATCH * NSEQ * D_MODEL;
  float* pres_v = pres_k + (size_t)BATCH * NHEAD * NSEQ * DHEAD;

  char* ws = (char*)d_ws;
  const size_t MB = 1024 * 1024;  // total workspace use: 40 MB
  __bf16* wWq     = (__bf16*)(ws + 0 * MB);
  __bf16* wWk     = (__bf16*)(ws + 2 * MB);
  __bf16* wWv     = (__bf16*)(ws + 4 * MB);
  __bf16* wWo     = (__bf16*)(ws + 6 * MB);
  __bf16* q_bf    = (__bf16*)(ws + 8 * MB);
  __bf16* k_bf    = (__bf16*)(ws + 16 * MB);
  __bf16* vT_bf   = (__bf16*)(ws + 24 * MB);
  __bf16* attn_bf = (__bf16*)(ws + 32 * MB);

  const int n4 = (D_MODEL * D_MODEL) / 4;
  k_cvt<<<(n4 + 255) / 256, 256, 0, stream>>>(Wq, wWq, n4);
  k_cvt<<<(n4 + 255) / 256, 256, 0, stream>>>(Wk, wWk, n4);
  k_cvt<<<(n4 + 255) / 256, 256, 0, stream>>>(Wv, wWv, n4);
  k_cvt<<<(n4 + 255) / 256, 256, 0, stream>>>(Wo, wWo, n4);

  k_proj<<<dim3(D_MODEL / 128, (BATCH * NSEQ) / 128, 3), 256, 0, stream>>>(
      Q, K, V, wWq, wWk, wWv, bq, bk, bv, q_bf, pres_k, k_bf, pres_v, vT_bf);

  k_attn<<<dim3(NSEQ / 128, BATCH * NHEAD), 256, 0, stream>>>(q_bf, k_bf, vT_bf,
                                                              attn_bf);

  k_out<<<dim3(D_MODEL / 128, (BATCH * NSEQ) / 128), 256, 0, stream>>>(attn_bf, wWo,
                                                                       bo, out);
}
